// Restoration_25039659335975
// MI455X (gfx1250) — matrix operand order repaired and checked by reading, not compiled, not run on hardware
//
#include <hip/hip_runtime.h>
#include <hip/hip_bf16.h>
#include <math.h>

// ---------------- problem constants ----------------
#define Bc   2
#define Cc   64
#define Hc   128
#define Wc   128
#define HWc  (Hc*Wc)          // 16384
#define Lc   HWc
#define DI   128              // D_INNER
#define NS   8                // D_STATE
#define DTR  4                // DT_RANK
#define CDB  20               // DT_RANK + 2*D_STATE
#define HIDc 192
#define NCH  128              // scan chunks
#define CHL  128              // chunk length (NCH*CHL == Lc)
#define EPSc 1e-5f

// ---------------- WMMA / TDM types & helpers ----------------
typedef __attribute__((ext_vector_type(16))) __bf16   v16bf;
typedef __attribute__((ext_vector_type(8)))  float    v8f;
typedef __attribute__((ext_vector_type(4)))  unsigned u32x4;
typedef __attribute__((ext_vector_type(8)))  int      i32x8;
typedef __attribute__((ext_vector_type(4)))  int      i32x4;

#if defined(__has_builtin)
#if __has_builtin(__builtin_amdgcn_tensor_load_to_lds)
#define HAVE_TDM 1
#endif
#endif
#ifndef HAVE_TDM
#define HAVE_TDM 0
#endif

__device__ __forceinline__ __bf16 f2bf(float f) {
  unsigned u = __float_as_uint(f);
  u += 0x7FFFu + ((u >> 16) & 1u);          // round-to-nearest-even
  unsigned short h = (unsigned short)(u >> 16);
  __bf16 b; __builtin_memcpy(&b, &h, sizeof(h)); return b;
}
__device__ __forceinline__ float bf2f(__bf16 b) {
  unsigned short h; __builtin_memcpy(&h, &b, sizeof(h));
  return __uint_as_float(((unsigned)h) << 16);
}
__device__ __forceinline__ float gelu_exact(float x) {
  return 0.5f * x * (1.f + erff(x * 0.70710678118654752f));
}

#if HAVE_TDM
// Build a D# for a flat 1-D copy of `nelem` 2-byte elements (global -> LDS)
// per CDNA5 ISA ch.8: group0 = {count/flags, lds_addr, global_addr, type=2},
// group1 = {data_size=2B, tensor_dim0=tile_dim0=nelem, dims1..2 unit, stride=nelem}.
// This toolchain exposes the 6-arg builtin: (g0, g1, g2, g3, g4, cpol).
__device__ __forceinline__ void tdm_copy_1d(const void* gsrc, unsigned lds_off,
                                            int nelem) {
  unsigned long long ga = (unsigned long long)(uintptr_t)gsrc;
  u32x4 g0;
  g0[0] = 1u;                                          // count=1, user D#
  g0[1] = lds_off;                                     // LDS byte address
  g0[2] = (unsigned)(ga & 0xFFFFFFFFull);              // global addr [31:0]
  g0[3] = (unsigned)((ga >> 32) & 0x01FFFFFFull) | (2u << 30);  // [56:32] | type=2
  i32x8 g1;
  g1[0] = (1 << 16);                                   // data_size=1 (2 bytes), wg_mask=0
  g1[1] = (int)((unsigned)(nelem & 0xFFFF) << 16);     // tensor_dim0 lo16 (bits 63:48)
  g1[2] = (int)((((unsigned)nelem >> 16) & 0xFFFFu) | (1u << 16)); // dim0 hi16 | tensor_dim1=1
  g1[3] = (int)((unsigned)(nelem & 0xFFFF) << 16);     // tile_dim0 (bits 127:112)
  g1[4] = 1;                                           // tile_dim1=1, tile_dim2=0
  g1[5] = nelem;                                       // tensor_dim0_stride lo32
  g1[6] = 0;
  g1[7] = 0;
  i32x4 z4 = (i32x4){0, 0, 0, 0};
  i32x8 z8 = (i32x8){0, 0, 0, 0, 0, 0, 0, 0};
  __builtin_amdgcn_tensor_load_to_lds(g0, g1, z4, z4, z8, 0);
}
#endif

// ---------------------------------------------------------------
// Split-precision bf16 WMMA GEMM:  D[rows,O] = A[rows,K] * W[O,K]^T
// Weights are pre-split into bf16 hi/lo arrays. Each block stages its
// 64-column weight slice in LDS via the Tensor Data Mover (wave 0 issues
// two tensor_load_to_lds, waits TENSORcnt, then a workgroup barrier).
// Each wave owns one 16-row tile and NT=4 col tiles; per 32-wide K step it
// builds the A hi/lo fragments once and issues 3 WMMAs per col tile
// (hi*hi + lo*hi + hi*lo) for ~f32 accuracy at bf16 matrix-op rate.
// ---------------------------------------------------------------
template<int K, int NT>
__global__ void gemm_bf16x3(const float* __restrict__ A,
                            const __bf16* __restrict__ Whi,
                            const __bf16* __restrict__ Wlo,
                            float* __restrict__ D,
                            int rows, int O)
{
  constexpr int SLICE = NT * 16 * K;          // bf16 elems per hi (or lo) slice
  __shared__ __bf16 sW[2 * SLICE];            // [0..SLICE) = hi, [SLICE..) = lo

  const int lane = threadIdx.x & 31;
  const int wv   = threadIdx.x >> 5;
  const int rowTile  = blockIdx.x * 8 + wv;
  const int colTile0 = blockIdx.y * NT;

#if HAVE_TDM
  if (threadIdx.x < 32) {
    tdm_copy_1d(Whi + (size_t)colTile0 * 16 * K,
                (unsigned)(uintptr_t)&sW[0], SLICE);
    tdm_copy_1d(Wlo + (size_t)colTile0 * 16 * K,
                (unsigned)(uintptr_t)&sW[SLICE], SLICE);
    __builtin_amdgcn_s_wait_tensorcnt((short)0);
  }
#else
  for (int i = threadIdx.x; i < SLICE; i += 256) {
    sW[i]         = Whi[(size_t)colTile0 * 16 * K + i];
    sW[SLICE + i] = Wlo[(size_t)colTile0 * 16 * K + i];
  }
#endif
  __syncthreads();

  const int m = lane & 15;                    // A row / output col owned by lane
  const int g = lane >> 4;                    // half-wave selector

  const float* Arow = A + (size_t)(rowTile * 16 + m) * K;

  v8f acc[NT];
  #pragma unroll
  for (int j = 0; j < NT; ++j) acc[j] = (v8f){0.f,0.f,0.f,0.f,0.f,0.f,0.f,0.f};

  #pragma unroll
  for (int k0 = 0; k0 < K; k0 += 32) {
    // ---- A fragment (built once, reused for NT col tiles) ----
    v16bf ah, al;
    #pragma unroll
    for (int e = 0; e < 16; ++e) {
      // A K order matches B: K = g*16+e (contiguous)
      int ka = g * 16 + e;
      float av = Arow[k0 + ka];
      __bf16 hi = f2bf(av);
      ah[e] = hi;
      al[e] = f2bf(av - bf2f(hi));
    }
    // ---- B fragments from LDS; 3 WMMAs per col tile ----
    #pragma unroll
    for (int j = 0; j < NT; ++j) {
      const int wrow = (j * 16 + m) * K + k0 + g * 16;  // K striped: lane group g
      v16bf bh, bl;
      #pragma unroll
      for (int e = 0; e < 16; ++e) {
        bh[e] = sW[wrow + e];
        bl[e] = sW[SLICE + wrow + e];
      }
      acc[j] = __builtin_amdgcn_wmma_f32_16x16x32_bf16(false, ah, false, bh, (short)0, acc[j], false, false);
      acc[j] = __builtin_amdgcn_wmma_f32_16x16x32_bf16(false, al, false, bh, (short)0, acc[j], false, false);
      acc[j] = __builtin_amdgcn_wmma_f32_16x16x32_bf16(false, ah, false, bl, (short)0, acc[j], false, false);
    }
  }

  // C/D layout: VGPR r -> M = r + 8*g, N = lane&15
  (void)rows;
  #pragma unroll
  for (int j = 0; j < NT; ++j) {
    #pragma unroll
    for (int r = 0; r < 8; ++r) {
      int mo = r + 8 * g;
      D[(size_t)(rowTile * 16 + mo) * O + (colTile0 + j) * 16 + m] = acc[j][r];
    }
  }
}

// Pre-split a weight matrix into bf16 hi/lo arrays (run once per launch).
__global__ void k_cvt_w(const float* __restrict__ W, __bf16* __restrict__ hi,
                        __bf16* __restrict__ lo, int n)
{
  int i = blockIdx.x * blockDim.x + threadIdx.x;
  if (i >= n) return;
  float v = W[i];
  __bf16 h = f2bf(v);
  hi[i] = h;
  lo[i] = f2bf(v - bf2f(h));
}

// ---------------------------------------------------------------
// K1: spatial flip + channel-LayerNorm. Writes xf (B,C,H,W) and
// the LN result transposed to (B*L, C) for the WMMA GEMM.
// ---------------------------------------------------------------
__global__ void k_flip_ln1(const float* __restrict__ x, const float* __restrict__ nw,
                           const float* __restrict__ nb, float* __restrict__ xf,
                           float* __restrict__ lnT)
{
  int t = blockIdx.x * blockDim.x + threadIdx.x;
  if (t >= Bc * HWc) return;
  int w = t % Wc, h = (t / Wc) % Hc, b = t / HWc;
  const float* src = x + (size_t)b * Cc * HWc + (size_t)(Hc - 1 - h) * Wc + (Wc - 1 - w);
  float s = 0.f, s2 = 0.f;
  for (int c = 0; c < Cc; ++c) { float v = src[(size_t)c * HWc]; s += v; s2 += v * v; }
  float mu = s * (1.f / Cc);
  float var = s2 * (1.f / Cc) - mu * mu;
  float ri = rsqrtf(var + EPSc);
  float* xo = xf + (size_t)b * Cc * HWc + (size_t)h * Wc + w;
  float* lo = lnT + (size_t)t * Cc;
  for (int c = 0; c < Cc; ++c) {
    float v = src[(size_t)c * HWc];
    xo[(size_t)c * HWc] = v;
    lo[c] = (v - mu) * ri * nw[c] + nb[c];
  }
}

// K3: depthwise 3x3 conv + bias + GELU on xi half of xz (B,L,256) -> xs (B,DI,L)
__global__ void k_dwconv2d_gelu(const float* __restrict__ xz, const float* __restrict__ cw,
                                const float* __restrict__ cb, float* __restrict__ xs)
{
  int t = blockIdx.x * blockDim.x + threadIdx.x;
  if (t >= Bc * DI * HWc) return;
  int d = t % DI; int w = (t / DI) % Wc; int h = (t / (DI * Wc)) % Hc; int b = t / (DI * HWc);
  float acc = cb[d];
  #pragma unroll
  for (int dh = -1; dh <= 1; ++dh) {
    int hh = h + dh; if (hh < 0 || hh >= Hc) continue;
    #pragma unroll
    for (int dw = -1; dw <= 1; ++dw) {
      int ww = w + dw; if (ww < 0 || ww >= Wc) continue;
      size_t row = (size_t)(b * Hc + hh) * Wc + ww;
      acc += xz[row * (2 * DI) + d] * cw[d * 9 + (dh + 1) * 3 + (dw + 1)];
    }
  }
  xs[(size_t)(b * DI + d) * Lc + (size_t)h * Wc + w] = gelu_exact(acc);
}

// K4: x_dbl[b,c,l] = sum_d x_proj_w[c,d] * xs[b,d,l]   (O=20 -> VALU)
__global__ void k_xproj(const float* __restrict__ xs, const float* __restrict__ pw,
                        float* __restrict__ xd)
{
  int t = blockIdx.x * blockDim.x + threadIdx.x;
  if (t >= Bc * CDB * Lc) return;
  int l = t % Lc; int c = (t / Lc) % CDB; int b = t / (CDB * Lc);
  float acc = 0.f;
  for (int d = 0; d < DI; ++d)
    acc += pw[c * DI + d] * xs[(size_t)(b * DI + d) * Lc + l];
  xd[(size_t)(b * CDB + c) * Lc + l] = acc;
}

// K5: depthwise conv1d (k=7, pad=3) + bias over L
__global__ void k_dwconv1d(const float* __restrict__ xd, const float* __restrict__ cw,
                           const float* __restrict__ cb, float* __restrict__ out)
{
  int t = blockIdx.x * blockDim.x + threadIdx.x;
  if (t >= Bc * CDB * Lc) return;
  int l = t % Lc; int c = (t / Lc) % CDB; int b = t / (CDB * Lc);
  float acc = cb[c];
  const float* src = xd + (size_t)(b * CDB + c) * Lc;
  #pragma unroll
  for (int i = 0; i < 7; ++i) {
    int ll = l + i - 3;
    if (ll >= 0 && ll < Lc) acc += src[ll] * cw[c * 7 + i];
  }
  out[(size_t)(b * CDB + c) * Lc + l] = acc;
}

// K6: delta = softplus(dt_w @ dts + dt_b)
__global__ void k_delta(const float* __restrict__ xdc, const float* __restrict__ dtw,
                        const float* __restrict__ dtb, float* __restrict__ dlt)
{
  int t = blockIdx.x * blockDim.x + threadIdx.x;
  if (t >= Bc * DI * Lc) return;
  int l = t % Lc; int d = (t / Lc) % DI; int b = t / (DI * Lc);
  float acc = dtb[d];
  #pragma unroll
  for (int r = 0; r < DTR; ++r)
    acc += dtw[d * DTR + r] * xdc[(size_t)(b * CDB + r) * Lc + l];
  dlt[(size_t)(b * DI + d) * Lc + l] = (acc > 20.f) ? acc : log1pf(__expf(acc));
}

// ---- chunked parallel linear-recurrence scan: h <- exp(dl*A)*h + dl*u*B ----
// Phase 1: per (b,d,chunk) compute chunk decay product P and zero-init partial H.
__global__ void k_scan_chunk(const float* __restrict__ dlt, const float* __restrict__ xs,
                             const float* __restrict__ xdc, const float* __restrict__ alog,
                             float* __restrict__ chP, float* __restrict__ chH)
{
  int t = blockIdx.x * blockDim.x + threadIdx.x;
  if (t >= Bc * DI * NCH) return;
  int c = t % NCH; int d = (t / NCH) % DI; int b = t / (DI * NCH);
  float Aa[NS];
  #pragma unroll
  for (int n = 0; n < NS; ++n) Aa[n] = -__expf(alog[d * NS + n]);
  float h[NS], P[NS];
  #pragma unroll
  for (int n = 0; n < NS; ++n) { h[n] = 0.f; P[n] = 1.f; }
  size_t base  = (size_t)(b * DI + d) * Lc;
  size_t bbase = (size_t)(b * CDB + DTR) * Lc;    // Bs rows
  for (int i = 0; i < CHL; ++i) {
    int l = c * CHL + i;
    float dl = dlt[base + l];
    float du = dl * xs[base + l];
    #pragma unroll
    for (int n = 0; n < NS; ++n) {
      float e  = __expf(dl * Aa[n]);
      float Bn = xdc[bbase + (size_t)n * Lc + l];
      P[n] *= e;
      h[n]  = e * h[n] + du * Bn;
    }
  }
  size_t o = (size_t)t * NS;
  #pragma unroll
  for (int n = 0; n < NS; ++n) { chP[o + n] = P[n]; chH[o + n] = h[n]; }
}

// Phase 2: sequential prefix across the 128 chunk summaries (2048 threads).
__global__ void k_scan_prefix(const float* __restrict__ chP, const float* __restrict__ chH,
                              float* __restrict__ hin)
{
  int t = blockIdx.x * blockDim.x + threadIdx.x;
  if (t >= Bc * DI * NS) return;
  int n = t % NS; int d = (t / NS) % DI; int b = t / (DI * NS);
  size_t base = (size_t)(b * DI + d) * NCH * NS + n;
  float run = 0.f;
  for (int c = 0; c < NCH; ++c) {
    size_t idx = base + (size_t)c * NS;
    hin[idx] = run;
    run = chP[idx] * run + chH[idx];
  }
}

// Phase 3: re-run each chunk from its correct initial state; emit y + Ds*u.
__global__ void k_scan_apply(const float* __restrict__ dlt, const float* __restrict__ xs,
                             const float* __restrict__ xdc, const float* __restrict__ alog,
                             const float* __restrict__ hin, const float* __restrict__ Dsp,
                             float* __restrict__ ys)
{
  int t = blockIdx.x * blockDim.x + threadIdx.x;
  if (t >= Bc * DI * NCH) return;
  int c = t % NCH; int d = (t / NCH) % DI; int b = t / (DI * NCH);
  float Aa[NS];
  #pragma unroll
  for (int n = 0; n < NS; ++n) Aa[n] = -__expf(alog[d * NS + n]);
  float h[NS];
  #pragma unroll
  for (int n = 0; n < NS; ++n) h[n] = hin[(size_t)t * NS + n];
  float Dd = Dsp[d];
  size_t base  = (size_t)(b * DI + d) * Lc;
  size_t bbase = (size_t)(b * CDB + DTR) * Lc;          // Bs
  size_t cbase = (size_t)(b * CDB + DTR + NS) * Lc;     // Cs
  for (int i = 0; i < CHL; ++i) {
    int l = c * CHL + i;
    float dl = dlt[base + l];
    float u  = xs[base + l];
    float du = dl * u;
    float y  = 0.f;
    #pragma unroll
    for (int n = 0; n < NS; ++n) {
      float e  = __expf(dl * Aa[n]);
      float Bn = xdc[bbase + (size_t)n * Lc + l];
      float Cn = xdc[cbase + (size_t)n * Lc + l];
      h[n] = e * h[n] + du * Bn;
      y += h[n] * Cn;
    }
    ys[base + l] = y + Dd * u;
  }
}

// K10: LN over D_INNER + gate with gelu(z); output (B*L, DI) for out_proj GEMM.
__global__ void k_outnorm_gate(const float* __restrict__ ys, const float* __restrict__ xz,
                               const float* __restrict__ ow, const float* __restrict__ ob,
                               float* __restrict__ yg)
{
  int t = blockIdx.x * blockDim.x + threadIdx.x;
  if (t >= Bc * Lc) return;
  int l = t % Lc; int b = t / Lc;
  float s = 0.f, s2 = 0.f;
  for (int d = 0; d < DI; ++d) {
    float v = ys[(size_t)(b * DI + d) * Lc + l];
    s += v; s2 += v * v;
  }
  float mu = s * (1.f / DI);
  float var = s2 * (1.f / DI) - mu * mu;
  float ri = rsqrtf(var + EPSc);
  for (int d = 0; d < DI; ++d) {
    float v  = ys[(size_t)(b * DI + d) * Lc + l];
    float yn = (v - mu) * ri * ow[d] + ob[d];
    float z  = xz[(size_t)t * (2 * DI) + DI + d];
    yg[(size_t)t * DI + d] = yn * gelu_exact(z);
  }
}

// K12: residual add + channel-LN #2. Writes x2 (B,C,H,W) and ln2T (B*L, C).
__global__ void k_resid_ln2(const float* __restrict__ xf, const float* __restrict__ prj,
                            const float* __restrict__ nw, const float* __restrict__ nb,
                            float* __restrict__ x2, float* __restrict__ lnT)
{
  int t = blockIdx.x * blockDim.x + threadIdx.x;
  if (t >= Bc * HWc) return;
  int l = t % HWc; int b = t / HWc;
  float s = 0.f, s2 = 0.f;
  for (int c = 0; c < Cc; ++c) {
    float v = xf[(size_t)(b * Cc + c) * HWc + l] + prj[(size_t)t * Cc + c];
    s += v; s2 += v * v;
  }
  float mu = s * (1.f / Cc);
  float var = s2 * (1.f / Cc) - mu * mu;
  float ri = rsqrtf(var + EPSc);
  for (int c = 0; c < Cc; ++c) {
    float v = xf[(size_t)(b * Cc + c) * HWc + l] + prj[(size_t)t * Cc + c];
    x2[(size_t)(b * Cc + c) * HWc + l] = v;
    lnT[(size_t)t * Cc + c] = (v - mu) * ri * nw[c] + nb[c];
  }
}

// K14: depthwise 3x3 (no bias) on 384 ch + GLU-style gate -> (B*L, 192)
__global__ void k_dwgate(const float* __restrict__ y1, const float* __restrict__ dww,
                         float* __restrict__ gt)
{
  int t = blockIdx.x * blockDim.x + threadIdx.x;
  if (t >= Bc * HIDc * HWc) return;
  int ch = t % HIDc; int w = (t / HIDc) % Wc; int h = (t / (HIDc * Wc)) % Hc; int b = t / (HIDc * HWc);
  float a = 0.f, bb = 0.f;
  #pragma unroll
  for (int dh = -1; dh <= 1; ++dh) {
    int hh = h + dh; if (hh < 0 || hh >= Hc) continue;
    #pragma unroll
    for (int dw = -1; dw <= 1; ++dw) {
      int ww = w + dw; if (ww < 0 || ww >= Wc) continue;
      size_t row = (size_t)(b * Hc + hh) * Wc + ww;
      int ki = (dh + 1) * 3 + (dw + 1);
      a  += y1[row * (2 * HIDc) + ch]         * dww[ch * 9 + ki];
      bb += y1[row * (2 * HIDc) + ch + HIDc]  * dww[(ch + HIDc) * 9 + ki];
    }
  }
  gt[((size_t)(b * Lc) + (size_t)h * Wc + w) * HIDc + ch] = gelu_exact(a) * bb;
}

// K16: per-8x8-patch rFFT2 * param -> irFFT2, plus final residual into d_out.
// Fully unrolled separable DFT; twiddles constant-fold to immediates.
__global__ void k_patchfft(const float* __restrict__ pv, const float* __restrict__ x2,
                           const float* __restrict__ fpar, float* __restrict__ out)
{
  int t = blockIdx.x * blockDim.x + threadIdx.x;
  if (t >= Bc * Cc * 16 * 16) return;
  int pw = t % 16; int ph = (t / 16) % 16; int c = (t / 256) % Cc; int b = t / (256 * Cc);

  const float CO[8] = {1.f, 0.70710678118654752f, 0.f, -0.70710678118654752f,
                       -1.f, -0.70710678118654752f, 0.f, 0.70710678118654752f};
  const float SI[8] = {0.f, 0.70710678118654752f, 1.f, 0.70710678118654752f,
                       0.f, -0.70710678118654752f, -1.f, -0.70710678118654752f};

  float o[64];
  #pragma unroll
  for (int i = 0; i < 64; ++i) o[i] = 0.f;

  #pragma unroll
  for (int v = 0; v < 5; ++v) {
    float gre[8], gim[8];
    #pragma unroll
    for (int y = 0; y < 8; ++y) {
      float re = 0.f, im = 0.f;
      #pragma unroll
      for (int x = 0; x < 8; ++x) {
        size_t row = (size_t)(b * Hc + ph * 8 + y) * Wc + pw * 8 + x;
        float p = pv[row * Cc + c];
        int k = (v * x) & 7;
        re += p * CO[k];
        im -= p * SI[k];
      }
      gre[y] = re; gim[y] = im;
    }
    float fre[8], fim[8];
    #pragma unroll
    for (int u = 0; u < 8; ++u) {
      float re = 0.f, im = 0.f;
      #pragma unroll
      for (int y = 0; y < 8; ++y) {
        int k = (u * y) & 7; float cc = CO[k], ss = SI[k];
        re += gre[y] * cc + gim[y] * ss;
        im += gim[y] * cc - gre[y] * ss;
      }
      float pm = fpar[c * 40 + u * 5 + v];
      fre[u] = re * pm; fim[u] = im * pm;
    }
    float wv = (v == 0 || v == 4) ? 1.f : 2.f;
    #pragma unroll
    for (int y = 0; y < 8; ++y) {
      float re = 0.f, im = 0.f;
      #pragma unroll
      for (int u = 0; u < 8; ++u) {
        int k = (u * y) & 7; float cc = CO[k], ss = SI[k];
        re += fre[u] * cc - fim[u] * ss;
        im += fre[u] * ss + fim[u] * cc;
      }
      re *= 0.125f; im *= 0.125f;
      #pragma unroll
      for (int x = 0; x < 8; ++x) {
        int k = (v * x) & 7;
        o[y * 8 + x] += wv * (re * CO[k] - im * SI[k]);
      }
    }
  }
  #pragma unroll
  for (int y = 0; y < 8; ++y) {
    #pragma unroll
    for (int x = 0; x < 8; ++x) {
      size_t idx = (size_t)(b * Cc + c) * HWc + (size_t)(ph * 8 + y) * Wc + pw * 8 + x;
      out[idx] = x2[idx] + 0.125f * o[y * 8 + x];
    }
  }
}

// ---------------------------------------------------------------
extern "C" void kernel_launch(void* const* d_in, const int* in_sizes, int n_in,
                              void* d_out, int out_size, void* d_ws, size_t ws_size,
                              hipStream_t stream)
{
  (void)in_sizes; (void)n_in; (void)out_size; (void)ws_size;
  const float* x    = (const float*)d_in[0];
  const float* n1w  = (const float*)d_in[1];
  const float* n1b  = (const float*)d_in[2];
  const float* inpw = (const float*)d_in[3];
  const float* c2w  = (const float*)d_in[4];
  const float* c2b  = (const float*)d_in[5];
  const float* xpw  = (const float*)d_in[6];
  const float* xcw  = (const float*)d_in[7];
  const float* xcb  = (const float*)d_in[8];
  const float* dtw  = (const float*)d_in[9];
  const float* dtb  = (const float*)d_in[10];
  const float* alog = (const float*)d_in[11];
  const float* Dsp  = (const float*)d_in[12];
  const float* onw  = (const float*)d_in[13];
  const float* onb  = (const float*)d_in[14];
  const float* opw  = (const float*)d_in[15];
  const float* n2w  = (const float*)d_in[16];
  const float* n2b  = (const float*)d_in[17];
  const float* pinw = (const float*)d_in[18];
  const float* dww  = (const float*)d_in[19];
  const float* fpar = (const float*)d_in[20];
  const float* potw = (const float*)d_in[21];

  float* ws = (float*)d_ws;
  size_t off = 0;
  float* XF  = ws + off; off += (size_t)Bc * Cc * HWc;        // flipped input
  float* LN  = ws + off; off += (size_t)Bc * HWc * Cc;        // LN1 / LN2 (B*L, C)
  size_t oXZ = off;
  float* XZ  = ws + off; off += (size_t)Bc * Lc * 2 * DI;     // in_proj out (B*L, 256)
  float* XS  = ws + off; off += (size_t)Bc * DI * Lc;         // conv+gelu (B,DI,L)
  float* XD  = ws + off; off += (size_t)Bc * CDB * Lc;
  float* XDC = ws + off; off += (size_t)Bc * CDB * Lc;
  size_t oDL = off;
  float* DLT = ws + off; off += (size_t)Bc * DI * Lc;
  float* CHP = ws + off; off += (size_t)Bc * DI * NCH * NS;
  float* CHH = ws + off; off += (size_t)Bc * DI * NCH * NS;
  float* HIN = ws + off; off += (size_t)Bc * DI * NCH * NS;
  float* YS  = ws + off; off += (size_t)Bc * DI * Lc;
  size_t oYG = off;
  float* YG  = ws + off; off += (size_t)Bc * Lc * DI;         // gated (B*L, DI)
  float* PRJ = ws + off; off += (size_t)Bc * Lc * Cc;         // out_proj (B*L, C)
  float* X2  = ws + off; off += (size_t)Bc * Cc * HWc;        // after residual 1
  // pre-split bf16 weights (hi/lo) appended after the float arena:
  __bf16* bfb = (__bf16*)(ws + off);
  size_t bo = 0;
  __bf16* IPH = bfb + bo; bo += 256 * 64;   __bf16* IPL = bfb + bo; bo += 256 * 64;
  __bf16* OPH = bfb + bo; bo += 64 * 128;   __bf16* OPL = bfb + bo; bo += 64 * 128;
  __bf16* PIH = bfb + bo; bo += 384 * 64;   __bf16* PIL = bfb + bo; bo += 384 * 64;
  __bf16* POH = bfb + bo; bo += 64 * 192;   __bf16* POL = bfb + bo; bo += 64 * 192;
  // phase-2 reuse of dead phase-1 regions:
  float* Y1  = ws + oXZ;   // pin out (B*L, 384)  <= XZ+XS  (exact fit)
  float* GT  = ws + oDL;   // gated (B*L, 192)    <= DLT..YS
  float* PV  = ws + oYG;   // pout out (B*L, 64)  <= YG

  const dim3 blk(256);
  const int rows = Bc * Lc;   // 32768

  // weight pre-split (tiny)
  k_cvt_w<<<dim3((256 * 64 + 255) / 256), blk, 0, stream>>>(inpw, IPH, IPL, 256 * 64);
  k_cvt_w<<<dim3((64 * 128 + 255) / 256), blk, 0, stream>>>(opw, OPH, OPL, 64 * 128);
  k_cvt_w<<<dim3((384 * 64 + 255) / 256), blk, 0, stream>>>(pinw, PIH, PIL, 384 * 64);
  k_cvt_w<<<dim3((64 * 192 + 255) / 256), blk, 0, stream>>>(potw, POH, POL, 64 * 192);

  // --- SS2D branch ---
  k_flip_ln1<<<dim3((Bc * HWc + 255) / 256), blk, 0, stream>>>(x, n1w, n1b, XF, LN);
  gemm_bf16x3<64, 4><<<dim3(rows / 128, 4), blk, 0, stream>>>(LN, IPH, IPL, XZ, rows, 2 * DI);
  k_dwconv2d_gelu<<<dim3((Bc * DI * HWc + 255) / 256), blk, 0, stream>>>(XZ, c2w, c2b, XS);
  k_xproj<<<dim3((Bc * CDB * Lc + 255) / 256), blk, 0, stream>>>(XS, xpw, XD);
  k_dwconv1d<<<dim3((Bc * CDB * Lc + 255) / 256), blk, 0, stream>>>(XD, xcw, xcb, XDC);
  k_delta<<<dim3((Bc * DI * Lc + 255) / 256), blk, 0, stream>>>(XDC, dtw, dtb, DLT);
  k_scan_chunk<<<dim3((Bc * DI * NCH + 255) / 256), blk, 0, stream>>>(DLT, XS, XDC, alog, CHP, CHH);
  k_scan_prefix<<<dim3((Bc * DI * NS + 255) / 256), blk, 0, stream>>>(CHP, CHH, HIN);
  k_scan_apply<<<dim3((Bc * DI * NCH + 255) / 256), blk, 0, stream>>>(DLT, XS, XDC, alog, HIN, Dsp, YS);
  k_outnorm_gate<<<dim3((Bc * Lc + 255) / 256), blk, 0, stream>>>(YS, XZ, onw, onb, YG);
  gemm_bf16x3<128, 4><<<dim3(rows / 128, 1), blk, 0, stream>>>(YG, OPH, OPL, PRJ, rows, Cc);
  k_resid_ln2<<<dim3((Bc * HWc + 255) / 256), blk, 0, stream>>>(XF, PRJ, n2w, n2b, X2, LN);

  // --- EDFFN branch ---
  gemm_bf16x3<64, 4><<<dim3(rows / 128, 6), blk, 0, stream>>>(LN, PIH, PIL, Y1, rows, 2 * HIDc);
  k_dwgate<<<dim3((Bc * HIDc * HWc + 255) / 256), blk, 0, stream>>>(Y1, dww, GT);
  gemm_bf16x3<192, 4><<<dim3(rows / 128, 1), blk, 0, stream>>>(GT, POH, POL, PV, rows, Cc);
  k_patchfft<<<dim3((Bc * Cc * 256 + 255) / 256), blk, 0, stream>>>(PV, X2, fpar, (float*)d_out);
}
